// WeaveModule_24068996727415
// MI455X (gfx1250) — compile-verified, matrix operand order checked by reading, not run on hardware
//
#include <hip/hip_runtime.h>

typedef __attribute__((ext_vector_type(16))) _Float16 v16h;
typedef __attribute__((ext_vector_type(8)))  float    v8f;
typedef __attribute__((ext_vector_type(4)))  float    f4;

namespace {

constexpr int TM   = 64;    // rows per workgroup
constexpr int NT   = 256;   // threads per workgroup (8 wave32)
constexpr int LDXP = 40;    // lds_x row pitch in halfs (32 + 8 pad -> conflict-free b128)
constexpr int KP1  = 136;   // W1 / lds_h pitch in halfs (128 + 8 pad)

union Frag {
  v16h h;
  f4   f[2];
};

__device__ __forceinline__ Frag load_frag(const _Float16* p) {
  Frag r;
  r.f[0] = *(const f4*)(p);
  r.f[1] = *(const f4*)(p + 16);
  return r;
}

__device__ __forceinline__ void async_cp16(const _Float16* lds_dst,
                                           const _Float16* gsrc) {
  const unsigned ldsa = (unsigned)(uintptr_t)lds_dst;
  const unsigned long long ga = (unsigned long long)gsrc;
  asm volatile("global_load_async_to_lds_b128 %0, %1, off"
               :: "v"(ldsa), "v"(ga) : "memory");
}

// ---------------------------------------------------------------------------
// One-shot weight prep: f32 [K,128] -> f16 transposed [128, KP] (zero padded),
// laid out exactly like the kernels' LDS so staging is a pure async DMA copy.
// ---------------------------------------------------------------------------
struct WTab {
  const float* src[12];
  _Float16*    dst[12];
  int          K0[12];
  int          KP[12];
};

__global__ __launch_bounds__(NT) void prep_weights(WTab t) {
  const int m = blockIdx.x;              // 0..11
  const float* __restrict__ src = t.src[m];
  _Float16* __restrict__ dst = t.dst[m];
  const int K0 = t.K0[m], KP = t.KP[m];
  for (int idx = threadIdx.x; idx < 128 * KP; idx += NT) {
    const int n = idx / KP;
    const int k = idx - n * KP;
    const float v = (k < K0) ? src[(long)k * 128 + n] : 0.f;
    dst[(long)n * KP + k] = (_Float16)v;
  }
}

// ---------------------------------------------------------------------------
// Fused y = relu( relu(X@W0+b0) @ W1 + b1 ), C = 128 output features.
// Weights come pre-transposed/padded f16; staged whole into LDS via async DMA.
// IN_MODE:  0 = f32 X [rows,K0] row-major
//           1 = gather: row (b,i,j) -> concat(atom_x[b,s0,:75], atom_x[b,s1,:75])
//               dual pass: pass0 (i,j), pass1 (j,i), summed after final relu
//           2 = f16 X [rows,256] (workspace concat) -> async DMA staging
// OUT_MODE: 0 = f16 out [rows,256] at column offset co
//           1 = f32 out [rows,128] (final output)
//           2 = f16: sum block's 64 rows -> out[blockIdx.x*256 + co + c]
// ---------------------------------------------------------------------------
template<int IN_MODE, int OUT_MODE, int K0>
__global__ __launch_bounds__(NT)
void fused_mlp2(const float* __restrict__ Xf, const _Float16* __restrict__ Xh,
                const float* __restrict__ atomx,
                const _Float16* __restrict__ W0T, const float* __restrict__ B0,
                const _Float16* __restrict__ W1T, const float* __restrict__ B1,
                float* __restrict__ outF, _Float16* __restrict__ outH,
                int co)
{
  constexpr int NPASS = (IN_MODE == 1) ? 2 : 1;
  constexpr int NKC   = (K0 + 31) >> 5;
  constexpr int KP0   = NKC * 32 + 8;          // W0 pitch in halfs

  __shared__ __align__(16) _Float16 lds_w0[128 * KP0]; // full W0^T (f16)
  __shared__ __align__(16) _Float16 lds_w1[128 * KP1]; // full W1^T (f16)
  __shared__ __align__(16) _Float16 lds_x[TM * LDXP];  // X chunk (64 x 32)
  __shared__ __align__(16) _Float16 lds_h[TM * KP1];   // layer-1 activations

  const int tid  = threadIdx.x;
  const int wave = tid >> 5;
  const int lane = tid & 31;
  const int lm   = lane & 15;
  const int hi   = lane >> 4;        // lane half -> K sub-base per WMMA A/B layout
  const int kb   = hi << 3;          // 0 or 8 halfs
  const int rp   = wave >> 2;        // row pair 0..1 -> 32 rows
  const int cq   = wave & 3;         // col quad 0..3 -> 32 cols
  const int arow0 = rp * 32 + lm;
  const int bcol0 = cq * 32 + lm;
  const long row0 = (long)blockIdx.x * TM;

  // ---- stage both weight matrices whole into LDS via async DMA (once) ----
  {
    constexpr int NG0 = (128 * KP0) / 8;   // 16B granules
    for (int g = tid; g < NG0; g += NT)
      async_cp16(lds_w0 + g * 8, W0T + g * 8);
    constexpr int NG1 = (128 * KP1) / 8;
    for (int g = tid; g < NG1; g += NT)
      async_cp16(lds_w1 + g * 8, W1T + g * 8);
  }

  float res[2][2][8];
#pragma unroll
  for (int ri = 0; ri < 2; ++ri)
#pragma unroll
    for (int ci = 0; ci < 2; ++ci)
#pragma unroll
      for (int g = 0; g < 8; ++g) res[ri][ci][g] = 0.f;

  for (int pass = 0; pass < NPASS; ++pass) {
    // ---------------- layer 1: h = relu(X @ W0 + b0) ----------------
    v8f acc[2][2];
#pragma unroll
    for (int ri = 0; ri < 2; ++ri)
#pragma unroll
      for (int ci = 0; ci < 2; ++ci)
#pragma unroll
        for (int g = 0; g < 8; ++g) acc[ri][ci][g] = 0.f;

#pragma unroll
    for (int kc = 0; kc < NKC; ++kc) {
      const int k0 = kc << 5;
      __syncthreads();
      // ---- stage X chunk [64 x 32] as f16 ----
      if (IN_MODE == 2) {
        // pure f16 copy: async DMA, one 16B granule per thread
        const int r = tid >> 2;
        const int q = tid & 3;
        async_cp16(lds_x + r * LDXP + q * 8,
                   Xh + (row0 + r) * 256 + k0 + q * 8);
      } else {
        // batched: issue all 8 loads, then convert+store (no per-elem waits)
        float xv[8];
#pragma unroll
        for (int u = 0; u < 8; ++u) {
          const int idx = tid + u * NT;       // TM*32 == 8*NT exactly
          const int r  = idx >> 5;
          const int kk = idx & 31;
          const int f  = k0 + kk;
          const long grow = row0 + r;
          float v = 0.f;
          if (IN_MODE == 0) {
            if (f < K0) v = Xf[grow * (long)K0 + f];
          } else { // gather
            const int g32 = (int)grow;
            const int b = g32 >> 12;
            const int i = (g32 >> 6) & 63;
            const int j = g32 & 63;
            const int s0 = pass ? j : i;
            const int s1 = pass ? i : j;
            if (f < 75)       v = atomx[((long)b * 64 + s0) * 75 + f];
            else if (f < 150) v = atomx[((long)b * 64 + s1) * 75 + (f - 75)];
          }
          xv[u] = v;
        }
#pragma unroll
        for (int u = 0; u < 8; ++u) {
          const int idx = tid + u * NT;
          lds_x[(idx >> 5) * LDXP + (idx & 31)] = (_Float16)xv[u];
        }
      }
      // first chunk's wait also covers the weight DMA issued above
      asm volatile("s_wait_asynccnt 0x0" ::: "memory");
      __syncthreads();

      Frag a[2], b[2];
#pragma unroll
      for (int ri = 0; ri < 2; ++ri)
        a[ri] = load_frag(&lds_x[(arow0 + 16 * ri) * LDXP + kb]);
#pragma unroll
      for (int ci = 0; ci < 2; ++ci)
        b[ci] = load_frag(&lds_w0[(bcol0 + 16 * ci) * KP0 + k0 + kb]);
#pragma unroll
      for (int ri = 0; ri < 2; ++ri)
#pragma unroll
        for (int ci = 0; ci < 2; ++ci)
          acc[ri][ci] = __builtin_amdgcn_wmma_f32_16x16x32_f16(
              false, a[ri].h, false, b[ci].h, (short)0, acc[ri][ci],
              false, false);
    }

    // bias + relu -> lds_h (f16), becomes layer-2 A matrix
#pragma unroll
    for (int ri = 0; ri < 2; ++ri) {
      const int rb = rp * 32 + ri * 16 + hi * 8;
#pragma unroll
      for (int ci = 0; ci < 2; ++ci) {
        const int n = bcol0 + 16 * ci;
        const float bias = B0[n];
#pragma unroll
        for (int g = 0; g < 8; ++g) {
          float h = acc[ri][ci][g] + bias;
          h = fmaxf(h, 0.f);
          lds_h[(rb + g) * KP1 + n] = (_Float16)h;
        }
      }
    }
    __syncthreads();

    // -------- layer 2: y = relu(h @ W1 + b1), K = 128, barrier-free --------
    v8f acc2[2][2];
#pragma unroll
    for (int ri = 0; ri < 2; ++ri)
#pragma unroll
      for (int ci = 0; ci < 2; ++ci)
#pragma unroll
        for (int g = 0; g < 8; ++g) acc2[ri][ci][g] = 0.f;

#pragma unroll
    for (int kc = 0; kc < 4; ++kc) {
      const int k0 = kc << 5;
      Frag a[2], b[2];
#pragma unroll
      for (int ri = 0; ri < 2; ++ri)
        a[ri] = load_frag(&lds_h[(arow0 + 16 * ri) * KP1 + k0 + kb]);
#pragma unroll
      for (int ci = 0; ci < 2; ++ci)
        b[ci] = load_frag(&lds_w1[(bcol0 + 16 * ci) * KP1 + k0 + kb]);
#pragma unroll
      for (int ri = 0; ri < 2; ++ri)
#pragma unroll
        for (int ci = 0; ci < 2; ++ci)
          acc2[ri][ci] = __builtin_amdgcn_wmma_f32_16x16x32_f16(
              false, a[ri].h, false, b[ci].h, (short)0, acc2[ri][ci],
              false, false);
    }

#pragma unroll
    for (int ri = 0; ri < 2; ++ri)
#pragma unroll
      for (int ci = 0; ci < 2; ++ci) {
        const int n = bcol0 + 16 * ci;
        const float bias = B1[n];
#pragma unroll
        for (int g = 0; g < 8; ++g)
          res[ri][ci][g] += fmaxf(acc2[ri][ci][g] + bias, 0.f);
      }
  }

  // ---------------- writeback ----------------
  if (OUT_MODE == 0) {
#pragma unroll
    for (int ri = 0; ri < 2; ++ri) {
      const int rb = rp * 32 + ri * 16 + hi * 8;
#pragma unroll
      for (int ci = 0; ci < 2; ++ci) {
        const int n = bcol0 + 16 * ci;
#pragma unroll
        for (int g = 0; g < 8; ++g)
          outH[(row0 + rb + g) * 256 + co + n] = (_Float16)res[ri][ci][g];
      }
    }
  } else if (OUT_MODE == 1) {
#pragma unroll
    for (int ri = 0; ri < 2; ++ri) {
      const int rb = rp * 32 + ri * 16 + hi * 8;
#pragma unroll
      for (int ci = 0; ci < 2; ++ci) {
        const int n = bcol0 + 16 * ci;
#pragma unroll
        for (int g = 0; g < 8; ++g)
          outF[(row0 + rb + g) * 128 + n] = res[ri][ci][g];
      }
    }
  } else {
    // sum this block's 64 rows (all j for fixed (b,i)) -> 128 values
    __shared__ float red[8 * 128];
#pragma unroll
    for (int ri = 0; ri < 2; ++ri)
#pragma unroll
      for (int ci = 0; ci < 2; ++ci) {
        float s = 0.f;
#pragma unroll
        for (int g = 0; g < 8; ++g) s += res[ri][ci][g];
        const int slot = (rp * 2 + ri) * 2 + hi;        // unique row-half 0..7
        red[slot * 128 + bcol0 + 16 * ci] = s;          // unique (slot, n)
      }
    __syncthreads();
    if (tid < 128) {
      float s = 0.f;
#pragma unroll
      for (int q = 0; q < 8; ++q) s += red[q * 128 + tid];
      outH[(long)blockIdx.x * 256 + co + tid] = (_Float16)s;
    }
  }
}

constexpr int KPof(int K0) { return ((K0 + 31) / 32) * 32 + 8; }

} // namespace

extern "C" void kernel_launch(void* const* d_in, const int* in_sizes, int n_in,
                              void* d_out, int out_size, void* d_ws, size_t ws_size,
                              hipStream_t stream)
{
  (void)in_sizes; (void)n_in; (void)out_size; (void)ws_size;

  const float* atom_x = (const float*)d_in[0];   // [64,64,75]
  const float* pair_x = (const float*)d_in[1];   // [64,4096,14]
  const float* W0s[6] = { (const float*)d_in[2],  (const float*)d_in[6],
                          (const float*)d_in[10], (const float*)d_in[14],
                          (const float*)d_in[18], (const float*)d_in[22] };
  const float* b0s[6] = { (const float*)d_in[3],  (const float*)d_in[7],
                          (const float*)d_in[11], (const float*)d_in[15],
                          (const float*)d_in[19], (const float*)d_in[23] };
  const float* W1s[6] = { (const float*)d_in[4],  (const float*)d_in[8],
                          (const float*)d_in[12], (const float*)d_in[16],
                          (const float*)d_in[20], (const float*)d_in[24] };
  const float* b1s[6] = { (const float*)d_in[5],  (const float*)d_in[9],
                          (const float*)d_in[13], (const float*)d_in[17],
                          (const float*)d_in[21], (const float*)d_in[25] };
  const int K0s[6] = { 75, 14, 256, 150, 14, 256 };  // a2a,p2a,al,a2p,p2p,pl

  // workspace (f16): atomcat [4096,256] | paircat [262144,256] | weight tiles
  _Float16* atomcat = (_Float16*)d_ws;
  _Float16* paircat = atomcat + (size_t)4096 * 256;
  _Float16* wcur    = paircat + (size_t)262144 * 256;

  WTab tab;
  _Float16* w0T[6];
  _Float16* w1T[6];
  for (int m = 0; m < 6; ++m) {
    const int kp0 = KPof(K0s[m]);
    w0T[m] = wcur;  wcur += (size_t)128 * kp0;
    w1T[m] = wcur;  wcur += (size_t)128 * KP1;
    tab.src[2 * m]     = W0s[m];  tab.dst[2 * m]     = w0T[m];
    tab.K0[2 * m]      = K0s[m];  tab.KP[2 * m]      = kp0;
    tab.src[2 * m + 1] = W1s[m];  tab.dst[2 * m + 1] = w1T[m];
    tab.K0[2 * m + 1]  = 128;     tab.KP[2 * m + 1]  = KP1;
  }

  float* next_atom = (float*)d_out;                        // [4096,128]
  float* next_pair = (float*)d_out + (size_t)4096 * 128;   // [262144,128]

  // 0) one-shot weight transpose/convert (tiny)
  prep_weights<<<dim3(12), NT, 0, stream>>>(tab);

  // 1) atom_to_atom: atom_x -> atomcat[:, 0:128]
  fused_mlp2<0, 0, 75><<<dim3(4096 / TM), NT, 0, stream>>>(
      atom_x, nullptr, nullptr, w0T[0], b0s[0], w1T[0], b1s[0],
      nullptr, atomcat, 0);
  // 2) pair_to_atom (+ sum over j): pair_x -> atomcat[:, 128:256]
  fused_mlp2<0, 2, 14><<<dim3(262144 / TM), NT, 0, stream>>>(
      pair_x, nullptr, nullptr, w0T[1], b0s[1], w1T[1], b1s[1],
      nullptr, atomcat, 128);
  // 3) atom_layer: atomcat -> next_atom (outer relu idempotent)
  fused_mlp2<2, 1, 256><<<dim3(4096 / TM), NT, 0, stream>>>(
      nullptr, atomcat, nullptr, w0T[2], b0s[2], w1T[2], b1s[2],
      next_atom, nullptr, 0);
  // 4) atom_to_pair, (i,j)+(j,i) dual pass -> paircat[:, 0:128]
  fused_mlp2<1, 0, 150><<<dim3(262144 / TM), NT, 0, stream>>>(
      nullptr, nullptr, atom_x, w0T[3], b0s[3], w1T[3], b1s[3],
      nullptr, paircat, 0);
  // 5) pair_to_pair: pair_x -> paircat[:, 128:256]
  fused_mlp2<0, 0, 14><<<dim3(262144 / TM), NT, 0, stream>>>(
      pair_x, nullptr, nullptr, w0T[4], b0s[4], w1T[4], b1s[4],
      nullptr, paircat, 128);
  // 6) pair_layer: paircat -> next_pair
  fused_mlp2<2, 1, 256><<<dim3(262144 / TM), NT, 0, stream>>>(
      nullptr, paircat, nullptr, w0T[5], b0s[5], w1T[5], b1s[5],
      next_pair, nullptr, 0);
}
